// TemporalAttention_51470888075369
// MI455X (gfx1250) — matrix operand order repaired and checked by reading, not compiled, not run on hardware
//
#include <hip/hip_runtime.h>

// ---------------------------------------------------------------------------
// Problem constants (from reference): B=16, T=128, N=50, D=512, 8 heads x 64
// ---------------------------------------------------------------------------
#define BB 16
#define TT 128
#define NN 50
#define DD 512
#define HH 8
#define DH 64
#define K3 (3 * DD)            // 1536 (concat X + TLE feature dim)
#define MTOK (BB * TT * NN)    // 102400 tokens

typedef __attribute__((ext_vector_type(16))) __bf16 v16bf;
typedef __attribute__((ext_vector_type(8)))  float  v8f;
typedef __attribute__((ext_vector_type(4)))  float  v4f;
typedef __attribute__((ext_vector_type(4)))  __bf16 v4bf;

union FragU { v16bf v; uint4 u[2]; };

// Load a 16x(bf16) WMMA fragment from two 16-byte chunks.
__device__ __forceinline__ v16bf load_frag2(const __bf16* p0, const __bf16* p1) {
  FragU f;
  f.u[0] = *(const uint4*)p0;
  f.u[1] = *(const uint4*)p1;
  return f.v;
}

__device__ __forceinline__ v8f wmma_bf16(v16bf a, v16bf b, v8f c) {
  // D = A(16x32) * B(32x16) + C, f32 accumulate
  return __builtin_amdgcn_wmma_f32_16x16x32_bf16(false, a, false, b, (short)0, c,
                                                 false, false);
}

// ---------------------------------------------------------------------------
// Kernel 1: weight pack.  dst[n*K + k] = (bf16) src[k*Ncols + n]
// (transpose so WMMA B-fragments read contiguous K runs per output column)
// ---------------------------------------------------------------------------
__global__ void pack_wt_kernel(const float* __restrict__ src,
                               __bf16* __restrict__ dst, int K, int Ncols) {
  int idx = blockIdx.x * blockDim.x + threadIdx.x;
  if (idx >= K * Ncols) return;
  int n = idx / K;
  int k = idx - n * K;
  dst[idx] = (__bf16)src[(size_t)k * Ncols + n];
}

// ---------------------------------------------------------------------------
// Kernel 2: fused QKV projection.  C[m, c] = relu(H[m,:] @ W[:, c] + bias)
//   A = H (fp32 from X / TLE, converted to bf16 on the fly)
//   B = Wt (bf16, [1536 out-cols x 1536 K], rows 0..511=Wq, 512..1023=Wk, ...)
// Outputs scattered into per-(b,n,head) layouts for the attention kernel:
//   Qb,Kb: [(b*N+n)*H+h][t][d]   Vb: [(b*N+n)*H+h][d][t]  (transposed!)
// ---------------------------------------------------------------------------
__global__ __launch_bounds__(256) void qkv_gemm_kernel(
    const float* __restrict__ X, const float* __restrict__ TLE,
    const __bf16* __restrict__ Wt,
    const float* __restrict__ bq, const float* __restrict__ bk,
    const float* __restrict__ bv,
    __bf16* __restrict__ Qb, __bf16* __restrict__ Kb, __bf16* __restrict__ Vb) {
  __shared__ __bf16 As[128][64];   // 16 KB: M-tile x K-panel
  __shared__ __bf16 Bs[128][64];   // 16 KB: N-tile x K-panel

  const int n0 = blockIdx.x * 128;
  const int m0 = blockIdx.y * 128;
  const int tid = threadIdx.x;
  const int lane = tid & 31;
  const int wave = tid >> 5;
  const int wm = wave >> 1;                // 0..3 -> 32-row band
  const int wn = wave & 1;                 // 0..1 -> 64-col band
  const int rl = lane & 15;
  const int kb8 = (lane >> 4) * 8;         // A-frag K split {0..7}/{8..15}
  const int kh16 = (lane >> 4) * 16;       // unused; B-frag uses kb8, same K split as A

  const v8f vzero = {0.f, 0.f, 0.f, 0.f, 0.f, 0.f, 0.f, 0.f};
  v8f acc[2][4];
#pragma unroll
  for (int i = 0; i < 2; i++)
#pragma unroll
    for (int j = 0; j < 4; j++) acc[i][j] = vzero;

  // Cooperative tile loaders: thread handles (row = tid>>1, 32-wide K half)
  const int arow = tid >> 1;
  const int ahalf = tid & 1;
  const int am = m0 + arow;                // token id, always < MTOK
  const int ab = am / (TT * NN);
  const int ar = am - ab * (TT * NN);
  const int at = ar / NN;
  const int an = ar - at * NN;
  const float* xrow = X + (size_t)((ab * TT + at) * NN + an) * DD;
  const float* tlerow = TLE + (size_t)((ab * TT + at) * NN + an) * (2 * DD);
  const __bf16* brow = Wt + (size_t)(n0 + arow) * K3;

  for (int k0 = 0; k0 < K3; k0 += 64) {
    __syncthreads();
    {
      const int kg = k0 + ahalf * 32;
      const float* src = (kg < DD) ? (xrow + kg) : (tlerow + (kg - DD));
#pragma unroll
      for (int i = 0; i < 32; i += 4) {
        v4f f = *(const v4f*)(src + i);
        *(v4bf*)&As[arow][ahalf * 32 + i] = __builtin_convertvector(f, v4bf);
      }
      const uint4* bs4 = (const uint4*)(brow + k0 + ahalf * 32);
      uint4* bd = (uint4*)&Bs[arow][ahalf * 32];
      bd[0] = bs4[0]; bd[1] = bs4[1]; bd[2] = bs4[2]; bd[3] = bs4[3];
      if (k0 + 64 < K3)  // prefetch next weight panel -> global_prefetch_b8
        __builtin_prefetch(brow + k0 + 64 + ahalf * 32, 0, 1);
    }
    __syncthreads();
#pragma unroll
    for (int kk = 0; kk < 64; kk += 32) {
      v16bf a[2];
#pragma unroll
      for (int mi = 0; mi < 2; mi++) {
        const __bf16* ap = &As[wm * 32 + mi * 16 + rl][kk + kb8];
        a[mi] = load_frag2(ap, ap + 16);
      }
#pragma unroll
      for (int f = 0; f < 4; f++) {
        const __bf16* bp = &Bs[wn * 64 + f * 16 + rl][kk + kb8];
        v16bf bm = load_frag2(bp, bp + 16);
#pragma unroll
        for (int mi = 0; mi < 2; mi++)
          acc[mi][f] = wmma_bf16(a[mi], bm, acc[mi][f]);
      }
    }
  }

  // Epilogue: bias + relu + bf16 scatter into head-major Q/K and d-major V.
  const int rh = lane >> 4;
#pragma unroll
  for (int mi = 0; mi < 2; mi++) {
#pragma unroll
    for (int f = 0; f < 4; f++) {
      const int c = n0 + wn * 64 + f * 16 + rl;  // global out column (0..1535)
      const int proj = c >> 9;                   // 0=q 1=k 2=v
      const int c9 = c & 511;
      const int hd = c9 >> 6;                    // head
      const int dd = c9 & 63;                    // dim in head
      const float bias = (proj == 0) ? bq[c9] : (proj == 1 ? bk[c9] : bv[c9]);
#pragma unroll
      for (int i = 0; i < 8; i++) {
        const int mrow = m0 + wm * 32 + mi * 16 + i + 8 * rh;
        const int b = mrow / (TT * NN);
        const int r = mrow - b * (TT * NN);
        const int t = r / NN;
        const int n = r - t * NN;
        float v = acc[mi][f][i] + bias;
        v = fmaxf(v, 0.0f);
        const __bf16 hv = (__bf16)v;
        const size_t hb = ((size_t)((b * NN + n) * HH + hd)) * (TT * DH);
        if (proj == 0)      Qb[hb + t * DH + dd] = hv;
        else if (proj == 1) Kb[hb + t * DH + dd] = hv;
        else                Vb[hb + dd * TT + t] = hv;  // transposed (d x t)
      }
    }
  }
}

// ---------------------------------------------------------------------------
// Kernel 3: attention for one (b, n, head).  256 threads = 8 waves.
// Wave w owns query rows [16w, 16w+16): S = Q K^T / 8, mask, softmax, O = P V.
// ---------------------------------------------------------------------------
__global__ __launch_bounds__(256) void attention_kernel(
    const __bf16* __restrict__ Qb, const __bf16* __restrict__ Kb,
    const __bf16* __restrict__ Vb, const int* __restrict__ kpm,
    __bf16* __restrict__ AO) {
  __shared__ __bf16 Ks[TT][DH];        // 16 KB  K  (t x d)
  __shared__ __bf16 Vs[DH][TT];        // 16 KB  V^T (d x t)
  __shared__ __bf16 Ps[8][16][TT];     // 32 KB  per-wave probabilities

  const int wg = blockIdx.x;                 // (b*NN + n)*HH + h
  const int hd = wg & (HH - 1);
  const int bn = wg >> 3;
  const int n = bn % NN;
  const int b = bn / NN;
  const size_t base = (size_t)wg * (TT * DH);

  const int tid = threadIdx.x;
  const int lane = tid & 31;
  const int wave = tid >> 5;
  const int rl = lane & 15;
  const int rh = lane >> 4;
  const int kb8 = rh * 8;
  const int kh16 = rh * 16;

  {  // stage K (each thread 64B)
    const int r = tid >> 1, hf = tid & 1;
    const uint4* src = (const uint4*)(Kb + base + r * DH + hf * 32);
    uint4* dst = (uint4*)&Ks[r][hf * 32];
    dst[0] = src[0]; dst[1] = src[1]; dst[2] = src[2]; dst[3] = src[3];
  }
  {  // stage V^T (each thread 64B)
    const int d = tid >> 2, q = tid & 3;
    const uint4* src = (const uint4*)(Vb + base + d * TT + q * 32);
    uint4* dst = (uint4*)&Vs[d][q * 32];
    dst[0] = src[0]; dst[1] = src[1]; dst[2] = src[2]; dst[3] = src[3];
  }
  __syncthreads();

  // Q fragments for this wave's 16 rows (A layout, straight from global)
  const __bf16* qrow = Qb + base + (size_t)(wave * 16 + rl) * DH;
  v16bf qa[2];
  qa[0] = load_frag2(qrow + kb8, qrow + kb8 + 16);
  qa[1] = load_frag2(qrow + 32 + kb8, qrow + 32 + kb8 + 16);

  const v8f vzero = {0.f, 0.f, 0.f, 0.f, 0.f, 0.f, 0.f, 0.f};
  v8f s[8];
#pragma unroll
  for (int f = 0; f < 8; f++) s[f] = vzero;

  // S = Q K^T : B-fragment column p = key index, K runs contiguous in Ks row
#pragma unroll
  for (int f = 0; f < 8; f++) {
#pragma unroll
    for (int kk = 0; kk < 64; kk += 32) {
      const __bf16* bp = &Ks[f * 16 + rl][kk + kb8];
      v16bf bm = load_frag2(bp, bp + 16);
      s[f] = wmma_bf16(qa[kk >> 5], bm, s[f]);
    }
  }

  // masked softmax over keys (row-wise; 16-lane shfl_xor reductions)
  const int kvlen = kpm[b];
#pragma unroll
  for (int i = 0; i < 8; i++) {
    const int qg = wave * 16 + i + 8 * rh;   // global query row
    float mx = -1e30f;
#pragma unroll
    for (int f = 0; f < 8; f++) {
      const int p = f * 16 + rl;             // key index
      float v = s[f][i] * 0.125f;            // 1/sqrt(64)
      v = ((p <= qg) && (p < kvlen)) ? v : -1e30f;
      s[f][i] = v;
      mx = fmaxf(mx, v);
    }
    mx = fmaxf(mx, __shfl_xor(mx, 1, 32));
    mx = fmaxf(mx, __shfl_xor(mx, 2, 32));
    mx = fmaxf(mx, __shfl_xor(mx, 4, 32));
    mx = fmaxf(mx, __shfl_xor(mx, 8, 32));
    float sum = 0.f;
#pragma unroll
    for (int f = 0; f < 8; f++) {
      float e = __expf(s[f][i] - mx);
      s[f][i] = e;
      sum += e;
    }
    sum += __shfl_xor(sum, 1, 32);
    sum += __shfl_xor(sum, 2, 32);
    sum += __shfl_xor(sum, 4, 32);
    sum += __shfl_xor(sum, 8, 32);
    const float inv = 1.0f / sum;
#pragma unroll
    for (int f = 0; f < 8; f++)
      Ps[wave][i + 8 * rh][f * 16 + rl] = (__bf16)(s[f][i] * inv);
  }
  __syncthreads();

  // O = P V : A = P (16 x 128 over 4 K-steps), B = V^T rows (contiguous K)
  v8f o[4];
#pragma unroll
  for (int f = 0; f < 4; f++) o[f] = vzero;
#pragma unroll
  for (int p0 = 0; p0 < TT; p0 += 32) {
    const __bf16* ap = &Ps[wave][rl][p0 + kb8];
    v16bf pa = load_frag2(ap, ap + 16);
#pragma unroll
    for (int f = 0; f < 4; f++) {
      const __bf16* bp = &Vs[f * 16 + rl][p0 + kb8];
      v16bf bm = load_frag2(bp, bp + 16);
      o[f] = wmma_bf16(pa, bm, o[f]);
    }
  }

  // store to token-major AO (bf16) for the FFN GEMMs
#pragma unroll
  for (int f = 0; f < 4; f++) {
#pragma unroll
    for (int i = 0; i < 8; i++) {
      const int t = wave * 16 + i + 8 * rh;
      const size_t row = (size_t)(b * TT + t) * NN + n;
      AO[row * DD + hd * DH + f * 16 + rl] = (__bf16)o[f][i];
    }
  }
}

// ---------------------------------------------------------------------------
// Kernels 4/5: bf16 GEMM, C = act(A @ Bt^T + bias).  A: [M x 512] bf16,
// Bt: [512 x 512] bf16 (pre-transposed).  RELU -> bf16 out; else f32 out.
// ---------------------------------------------------------------------------
template <int RELU, int OUTF32>
__global__ __launch_bounds__(256) void ffn_gemm_kernel(
    const __bf16* __restrict__ A, const __bf16* __restrict__ Bt,
    const float* __restrict__ bias, __bf16* __restrict__ obf,
    float* __restrict__ of32) {
  __shared__ __bf16 As[128][64];
  __shared__ __bf16 Bs[128][64];

  const int n0 = blockIdx.x * 128;
  const int m0 = blockIdx.y * 128;
  const int tid = threadIdx.x;
  const int lane = tid & 31;
  const int wave = tid >> 5;
  const int wm = wave >> 1;
  const int wn = wave & 1;
  const int rl = lane & 15;
  const int kb8 = (lane >> 4) * 8;
  const int kh16 = (lane >> 4) * 16;

  const v8f vzero = {0.f, 0.f, 0.f, 0.f, 0.f, 0.f, 0.f, 0.f};
  v8f acc[2][4];
#pragma unroll
  for (int i = 0; i < 2; i++)
#pragma unroll
    for (int j = 0; j < 4; j++) acc[i][j] = vzero;

  const int arow = tid >> 1;
  const int ahalf = tid & 1;
  const __bf16* arowp = A + (size_t)(m0 + arow) * DD;
  const __bf16* browp = Bt + (size_t)(n0 + arow) * DD;

  for (int k0 = 0; k0 < DD; k0 += 64) {
    __syncthreads();
    {
      const uint4* as4 = (const uint4*)(arowp + k0 + ahalf * 32);
      uint4* ad = (uint4*)&As[arow][ahalf * 32];
      ad[0] = as4[0]; ad[1] = as4[1]; ad[2] = as4[2]; ad[3] = as4[3];
      const uint4* bs4 = (const uint4*)(browp + k0 + ahalf * 32);
      uint4* bd = (uint4*)&Bs[arow][ahalf * 32];
      bd[0] = bs4[0]; bd[1] = bs4[1]; bd[2] = bs4[2]; bd[3] = bs4[3];
      if (k0 + 64 < DD) {
        __builtin_prefetch(arowp + k0 + 64 + ahalf * 32, 0, 1);
        __builtin_prefetch(browp + k0 + 64 + ahalf * 32, 0, 1);
      }
    }
    __syncthreads();
#pragma unroll
    for (int kk = 0; kk < 64; kk += 32) {
      v16bf a[2];
#pragma unroll
      for (int mi = 0; mi < 2; mi++) {
        const __bf16* ap = &As[wm * 32 + mi * 16 + rl][kk + kb8];
        a[mi] = load_frag2(ap, ap + 16);
      }
#pragma unroll
      for (int f = 0; f < 4; f++) {
        const __bf16* bp = &Bs[wn * 64 + f * 16 + rl][kk + kb8];
        v16bf bm = load_frag2(bp, bp + 16);
#pragma unroll
        for (int mi = 0; mi < 2; mi++)
          acc[mi][f] = wmma_bf16(a[mi], bm, acc[mi][f]);
      }
    }
  }

  const int rh = lane >> 4;
#pragma unroll
  for (int mi = 0; mi < 2; mi++) {
#pragma unroll
    for (int f = 0; f < 4; f++) {
      const int c = n0 + wn * 64 + f * 16 + rl;
      const float bi = bias[c];
#pragma unroll
      for (int i = 0; i < 8; i++) {
        const int mrow = m0 + wm * 32 + mi * 16 + i + 8 * rh;
        float v = acc[mi][f][i] + bi;
        if (RELU) v = fmaxf(v, 0.0f);
        if (OUTF32) of32[(size_t)mrow * DD + c] = v;
        else        obf[(size_t)mrow * DD + c] = (__bf16)v;
      }
    }
  }
}

// ---------------------------------------------------------------------------
// Host launcher
// ---------------------------------------------------------------------------
extern "C" void kernel_launch(void* const* d_in, const int* in_sizes, int n_in,
                              void* d_out, int out_size, void* d_ws,
                              size_t ws_size, hipStream_t stream) {
  (void)in_sizes; (void)n_in; (void)out_size; (void)ws_size;
  const float* X   = (const float*)d_in[0];
  const float* TLE = (const float*)d_in[1];
  const int*   kpm = (const int*)d_in[2];
  const float* Wq = (const float*)d_in[3];  const float* bq = (const float*)d_in[4];
  const float* Wk = (const float*)d_in[5];  const float* bk = (const float*)d_in[6];
  const float* Wv = (const float*)d_in[7];  const float* bv = (const float*)d_in[8];
  const float* W1 = (const float*)d_in[9];  const float* b1 = (const float*)d_in[10];
  const float* W2 = (const float*)d_in[11]; const float* b2 = (const float*)d_in[12];
  float* out = (float*)d_out;

  char* p = (char*)d_ws;
  auto take = [&](size_t bytes) {
    char* r = p;
    p += (bytes + 255) & ~(size_t)255;
    return r;
  };
  __bf16* Wt  = (__bf16*)take((size_t)K3 * K3 * 2);       // 4.7 MB
  __bf16* W1t = (__bf16*)take((size_t)DD * DD * 2);
  __bf16* W2t = (__bf16*)take((size_t)DD * DD * 2);
  __bf16* Qb  = (__bf16*)take((size_t)MTOK * DD * 2);     // 105 MB each
  __bf16* Kb  = (__bf16*)take((size_t)MTOK * DD * 2);
  __bf16* Vb  = (__bf16*)take((size_t)MTOK * DD * 2);
  __bf16* AO  = (__bf16*)take((size_t)MTOK * DD * 2);
  __bf16* Y1  = (__bf16*)take((size_t)MTOK * DD * 2);

  // 1) pack weights (transpose + bf16)
  pack_wt_kernel<<<(K3 * DD + 255) / 256, 256, 0, stream>>>(Wq, Wt, K3, DD);
  pack_wt_kernel<<<(K3 * DD + 255) / 256, 256, 0, stream>>>(
      Wk, Wt + (size_t)DD * K3, K3, DD);
  pack_wt_kernel<<<(K3 * DD + 255) / 256, 256, 0, stream>>>(
      Wv, Wt + (size_t)2 * DD * K3, K3, DD);
  pack_wt_kernel<<<(DD * DD + 255) / 256, 256, 0, stream>>>(W1, W1t, DD, DD);
  pack_wt_kernel<<<(DD * DD + 255) / 256, 256, 0, stream>>>(W2, W2t, DD, DD);

  // 2) fused QKV projection (relu) -> bf16 head-major buffers
  qkv_gemm_kernel<<<dim3(K3 / 128, MTOK / 128), 256, 0, stream>>>(
      X, TLE, Wt, bq, bk, bv, Qb, Kb, Vb);

  // 3) per-(b,n,head) causal attention
  attention_kernel<<<BB * NN * HH, 256, 0, stream>>>(Qb, Kb, Vb, kpm, AO);

  // 4) FFN1 (relu, bf16 out), 5) FFN2 (f32 out = final result)
  ffn_gemm_kernel<1, 0><<<dim3(DD / 128, MTOK / 128), 256, 0, stream>>>(
      AO, W1t, b1, Y1, nullptr);
  ffn_gemm_kernel<0, 1><<<dim3(DD / 128, MTOK / 128), 256, 0, stream>>>(
      Y1, W2t, b2, nullptr, out);
}
